// LinearTransformerEncoder_42691974922501
// MI455X (gfx1250) — matrix operand order repaired and checked by reading, not compiled, not run on hardware
//
#include <hip/hip_runtime.h>
#include <hip/hip_bf16.h>
#include <math.h>

// ---------------------------------------------------------------------------
// LinearTransformerEncoder on gfx1250 (MI455X).
// Weights are converted fp32->bf16 and transposed to [N][K] once per launch;
// GEMM-feeding activations get bf16 shadows written by producer epilogues.
// The WMMA GEMM hot loop is pure data movement: async global->LDS
// (triple buffered, 2 tiles in flight, partial ASYNCcnt waits) feeding
// v_wmma_f32_16x16x32_bf16 with fp32 accumulation.
// ---------------------------------------------------------------------------

#define BDIM 4
#define SDIM 2048
#define DDIM 1024
#define HDIM 16
#define LDIM 8
#define FDIM 4096
#define DH   64
#define BS   (BDIM * SDIM)        // 8192 rows

typedef __bf16 bf16x16 __attribute__((ext_vector_type(16)));
typedef __bf16 bf16x8  __attribute__((ext_vector_type(8)));
typedef __bf16 bf16x2  __attribute__((ext_vector_type(2)));
typedef float  f32x8   __attribute__((ext_vector_type(8)));
typedef int    v4i     __attribute__((ext_vector_type(4)));

union BF16Frag { bf16x16 v; bf16x8 h[2]; };

// ---- fp32 -> bf16 (RNE) helpers -------------------------------------------
__device__ __forceinline__ unsigned short f2bf_s(float f) {
  union { float f; unsigned int u; } x; x.f = f;
  return (unsigned short)((x.u + 0x7FFFu + ((x.u >> 16) & 1u)) >> 16);
}
__device__ __forceinline__ unsigned int pack_bf16(float lo, float hi) {
#if __has_builtin(__builtin_amdgcn_cvt_pk_bf16_f32)
  union { bf16x2 v; unsigned int u; } c;
  c.v = __builtin_amdgcn_cvt_pk_bf16_f32(lo, hi);
  return c.u;
#else
  union { float f; unsigned int u; } a, b; a.f = lo; b.f = hi;
  const unsigned int ra = a.u + 0x7FFFu + ((a.u >> 16) & 1u);
  const unsigned int rb = b.u + 0x7FFFu + ((b.u >> 16) & 1u);
  return __builtin_amdgcn_perm(rb, ra, 0x07060302u);
#endif
}

__device__ __forceinline__ float act_phi(float x) {   // elu(x)+1
  return x > 0.0f ? x + 1.0f : __expf(x);
}
__device__ __forceinline__ float act_gelu(float x) {  // exact gelu
  return 0.5f * x * (1.0f + erff(x * 0.70710678118654752f));
}

// ---- async global->LDS (CDNA5) with safe fallback -------------------------
#if __has_builtin(__builtin_amdgcn_global_load_async_to_lds_b128)
#define HAVE_ASYNC_LDS 1
__device__ __forceinline__ void async_cp16(const unsigned short* g, unsigned short* l) {
  __builtin_amdgcn_global_load_async_to_lds_b128(
      (__attribute__((address_space(1))) v4i*)(g),
      (__attribute__((address_space(3))) v4i*)(l), 0, 0);
}
template <int N>
__device__ __forceinline__ void wait_async() {
#if __has_builtin(__builtin_amdgcn_s_wait_asynccnt)
  __builtin_amdgcn_s_wait_asynccnt(N);
#else
  asm volatile("s_wait_asynccnt %0" :: "i"(N) : "memory");
#endif
}
#else
template <int N>
__device__ __forceinline__ void wait_async() {}
#endif

// ---------------------------------------------------------------------------
// Tiled bf16 WMMA GEMM: C[M,N] = epilogue(A[M,K] @ Bt[N,K]^T + bias[N]) (+res)
// A: bf16 row-major [M,K]; Bt: bf16 transposed weights [N,K].
// Block: 256 threads (8 waves). Tile 128x128, K-step 32.
// Triple-buffered LDS; 2 tiles of async copies in flight (4 asyncs/thread
// per tile, in-order completion -> wait asynccnt<=4 retires oldest tile).
// Wave grid 4(m) x 2(n); each wave owns 32x64 = 2x4 tiles of 16x16.
// ---------------------------------------------------------------------------
#define BK      32
#define LDS_A   40   // 32 + 8 pad halfwords (80B rows, 16B aligned)
#define LDS_B   40

template <int ACT, int RES, int OUTBF>
__global__ __launch_bounds__(256) void gemm_bf16_wmma(
    void* __restrict__ Cv, const unsigned short* __restrict__ A,
    const unsigned short* __restrict__ Bt, const float* __restrict__ bias,
    const float* __restrict__ res, int M, int N, int K) {

  __shared__ __align__(16) unsigned short sA[3][128 * LDS_A];
  __shared__ __align__(16) unsigned short sB[3][128 * LDS_B];

  const int tid  = threadIdx.x;
  const int lane = tid & 31;
  const int wid  = tid >> 5;
  const int wm   = wid & 3;   // 0..3 -> 32-row strip
  const int wn   = wid >> 2;  // 0..1 -> 64-col strip
  const int lm   = lane & 15;
  const int bm0  = blockIdx.y * 128;
  const int bn0  = blockIdx.x * 128;

  // staging: each thread copies 2x16B for A and 2x16B for B per K-step
  const int srow = tid >> 2;        // 0..63 (+64 on second pass)
  const int schk = (tid & 3) * 8;   // halfword offset within 32-wide row

  f32x8 acc[2][4];
#pragma unroll
  for (int mt = 0; mt < 2; ++mt)
#pragma unroll
    for (int nt = 0; nt < 4; ++nt) acc[mt][nt] = (f32x8)0.0f;

  const int akb = (lane < 16) ? 0 : 8;    // A frag K base
  const int bkb = (lane < 16) ? 0 : 8;    // B frag K base

  // stage tile at k0 into LDS buffer `buf` (4 async copies per thread)
  auto stage = [&](int k0, int buf) {
#pragma unroll
    for (int p = 0; p < 2; ++p) {
      const int r = srow + p * 64;
      const unsigned short* ga = &A [(size_t)(bm0 + r) * K + k0 + schk];
      const unsigned short* gb = &Bt[(size_t)(bn0 + r) * K + k0 + schk];
      unsigned short* la = &sA[buf][r * LDS_A + schk];
      unsigned short* lb = &sB[buf][r * LDS_B + schk];
#ifdef HAVE_ASYNC_LDS
      async_cp16(ga, la);
      async_cp16(gb, lb);
#else
      *(uint4*)la = *(const uint4*)ga;
      *(uint4*)lb = *(const uint4*)gb;
#endif
    }
  };

  const int NT = K / BK;
  stage(0, 0);
  if (NT > 1) stage(BK, 1);

  for (int kt = 0; kt < NT; ++kt) {
    // retire the oldest tile; keep the newer one in flight
    if (kt + 1 < NT) wait_async<4>();
    else             wait_async<0>();
    __syncthreads();

    const int cur = kt % 3;

    // ---- fragment loads from current buffer
    BF16Frag afrag[2], bfrag[4];
#pragma unroll
    for (int mt = 0; mt < 2; ++mt) {
      const int row = wm * 32 + mt * 16 + lm;
      afrag[mt].h[0] = *(const bf16x8*)&sA[cur][row * LDS_A + akb];
      afrag[mt].h[1] = *(const bf16x8*)&sA[cur][row * LDS_A + akb + 16];
    }
#pragma unroll
    for (int nt = 0; nt < 4; ++nt) {
      const int col = wn * 64 + nt * 16 + lm;
      bfrag[nt].h[0] = *(const bf16x8*)&sB[cur][col * LDS_B + bkb];
      bfrag[nt].h[1] = *(const bf16x8*)&sB[cur][col * LDS_B + bkb + 16];
    }

    // ---- prefetch tile kt+2 (two K-steps ahead) into the free buffer
    if (kt + 2 < NT) stage((kt + 2) * BK, (kt + 2) % 3);

#pragma unroll
    for (int mt = 0; mt < 2; ++mt)
#pragma unroll
      for (int nt = 0; nt < 4; ++nt)
        acc[mt][nt] = __builtin_amdgcn_wmma_f32_16x16x32_bf16(
            false, afrag[mt].v, false, bfrag[nt].v,
            (short)0, acc[mt][nt], false, false);
  }

  // ---- epilogue: bias + activation (+ residual); fp32 or bf16 output
  float* Cf = (float*)Cv;
  unsigned short* Cb = (unsigned short*)Cv;
#pragma unroll
  for (int mt = 0; mt < 2; ++mt) {
    const int base_row = bm0 + wm * 32 + mt * 16 + ((lane & 16) ? 8 : 0);
#pragma unroll
    for (int nt = 0; nt < 4; ++nt) {
      const int gcol = bn0 + wn * 64 + nt * 16 + lm;
      const float bval = bias[gcol];
#pragma unroll
      for (int r = 0; r < 8; ++r) {
        float v = acc[mt][nt][r] + bval;
        if (ACT == 1) v = act_phi(v);
        else if (ACT == 2) v = act_gelu(v);
        const size_t idx = (size_t)(base_row + r) * N + gcol;
        if (RES) v += res[idx];
        if (OUTBF) Cb[idx] = f2bf_s(v);
        else       Cf[idx] = v;
      }
    }
  }
}

// ---------------------------------------------------------------------------
// Weight prep: Wt[N,K] (bf16) = W[K,N] (fp32) transposed. 32x32 LDS tiles.
// grid = (N/32, K/32), block = 256.
// ---------------------------------------------------------------------------
__global__ __launch_bounds__(256) void transpose_bf16_kernel(
    const float* __restrict__ W, unsigned short* __restrict__ Wt, int K, int N) {
  __shared__ unsigned short ts[32][33];
  const int tid = threadIdx.x;
  const int n0 = blockIdx.x * 32, k0 = blockIdx.y * 32;
  const int tx = tid & 31, ty = tid >> 5;
#pragma unroll
  for (int j = 0; j < 4; ++j) {
    const int kk = ty + j * 8;
    ts[kk][tx] = f2bf_s(W[(size_t)(k0 + kk) * N + n0 + tx]);
  }
  __syncthreads();
  const int sx = (tid & 15) * 2, sy = tid >> 4;
#pragma unroll
  for (int j = 0; j < 2; ++j) {
    const int nn = sy + j * 16;
    const unsigned int p = (unsigned int)ts[sx][nn] |
                           ((unsigned int)ts[sx + 1][nn] << 16);
    *(unsigned int*)&Wt[(size_t)(n0 + nn) * K + k0 + sx] = p;
  }
}

// ---------------------------------------------------------------------------
// h = x (fp32 copy) and hb = bf16(x); grid = BS*D/1024, block 256
// ---------------------------------------------------------------------------
__global__ __launch_bounds__(256) void dualcast_kernel(
    const float* __restrict__ x, float* __restrict__ h,
    unsigned short* __restrict__ hb) {
  const size_t i = ((size_t)blockIdx.x * 256 + threadIdx.x) * 4;
  const float4 v = *(const float4*)&x[i];
  *(float4*)&h[i] = v;
  uint2 p; p.x = pack_bf16(v.x, v.y); p.y = pack_bf16(v.z, v.w);
  *(uint2*)&hb[i] = p;
}

// ---------------------------------------------------------------------------
// kv[b,h] = sum_s k[b,s,h,:]^T v[b,s,h,:]   (64x64),  ksum[b,h] = sum_s k
// ---------------------------------------------------------------------------
__global__ __launch_bounds__(256) void kv_ksum_kernel(
    const float* __restrict__ k, const float* __restrict__ v,
    float* __restrict__ kv, float* __restrict__ ksum) {
  const int bh = blockIdx.x;
  const int b = bh / HDIM, h = bh % HDIM;
  const int t = threadIdx.x;
  const int d  = t >> 2;
  const int m0 = (t & 3) * 16;

  const float* kb = k + (size_t)b * SDIM * DDIM + h * DH;
  const float* vb = v + (size_t)b * SDIM * DDIM + h * DH;

  float acc[16];
#pragma unroll
  for (int j = 0; j < 16; ++j) acc[j] = 0.0f;
  float ks = 0.0f;

  for (int s = 0; s < SDIM; ++s) {
    const float kd = kb[(size_t)s * DDIM + d];
    ks += kd;
    const float4 v0 = *(const float4*)&vb[(size_t)s * DDIM + m0 + 0];
    const float4 v1 = *(const float4*)&vb[(size_t)s * DDIM + m0 + 4];
    const float4 v2 = *(const float4*)&vb[(size_t)s * DDIM + m0 + 8];
    const float4 v3 = *(const float4*)&vb[(size_t)s * DDIM + m0 + 12];
    acc[0]  += kd * v0.x; acc[1]  += kd * v0.y; acc[2]  += kd * v0.z; acc[3]  += kd * v0.w;
    acc[4]  += kd * v1.x; acc[5]  += kd * v1.y; acc[6]  += kd * v1.z; acc[7]  += kd * v1.w;
    acc[8]  += kd * v2.x; acc[9]  += kd * v2.y; acc[10] += kd * v2.z; acc[11] += kd * v2.w;
    acc[12] += kd * v3.x; acc[13] += kd * v3.y; acc[14] += kd * v3.z; acc[15] += kd * v3.w;
  }
#pragma unroll
  for (int j = 0; j < 16; ++j)
    kv[(size_t)bh * (DH * DH) + d * DH + m0 + j] = acc[j];
  if ((t & 3) == 0) ksum[(size_t)bh * DH + d] = ks;
}

// ---------------------------------------------------------------------------
// att(bf16)[b,s,h,m] = (q . kv[:,m]) / (q . ksum + 1e-6)
// ---------------------------------------------------------------------------
__global__ __launch_bounds__(256) void attn_apply_kernel(
    const float* __restrict__ q, const float* __restrict__ kv,
    const float* __restrict__ ksum, unsigned short* __restrict__ attb) {
  const int s4 = blockIdx.x % (SDIM / 4);
  const int bh = blockIdx.x / (SDIM / 4);
  const int b = bh / HDIM, h = bh % HDIM;
  const int t = threadIdx.x;
  const int m = t & 63;
  const int s = s4 * 4 + (t >> 6);

  const float* qrow = q + (size_t)(b * SDIM + s) * DDIM + h * DH;
  const float* kvp  = kv + (size_t)bh * (DH * DH);
  const float* ksp  = ksum + (size_t)bh * DH;

  float num = 0.0f, den = 0.0f;
#pragma unroll 8
  for (int d = 0; d < DH; ++d) {
    const float qd = qrow[d];
    num += qd * kvp[d * DH + m];
    den += qd * ksp[d];
  }
  attb[(size_t)(b * SDIM + s) * DDIM + h * DH + m] = f2bf_s(num / (den + 1e-6f));
}

// ---------------------------------------------------------------------------
// LayerNorm over D=1024 (+ optional bf16 shadow output)
// ---------------------------------------------------------------------------
__global__ __launch_bounds__(256) void ln_kernel(
    float* __restrict__ out, const float* __restrict__ in,
    const float* __restrict__ g, const float* __restrict__ be,
    unsigned short* __restrict__ outb) {
  __shared__ float red[256];
  const int row = blockIdx.x;
  const int t = threadIdx.x;
  const float4 x = *(const float4*)&in[(size_t)row * DDIM + t * 4];

  red[t] = x.x + x.y + x.z + x.w;
  __syncthreads();
  for (int o = 128; o > 0; o >>= 1) {
    if (t < o) red[t] += red[t + o];
    __syncthreads();
  }
  const float mean = red[0] * (1.0f / DDIM);
  __syncthreads();

  const float dx = x.x - mean, dy = x.y - mean, dz = x.z - mean, dw = x.w - mean;
  red[t] = dx * dx + dy * dy + dz * dz + dw * dw;
  __syncthreads();
  for (int o = 128; o > 0; o >>= 1) {
    if (t < o) red[t] += red[t + o];
    __syncthreads();
  }
  const float inv = rsqrtf(red[0] * (1.0f / DDIM) + 1e-5f);

  const float4 gg = *(const float4*)&g[t * 4];
  const float4 bb = *(const float4*)&be[t * 4];
  float4 o4;
  o4.x = dx * inv * gg.x + bb.x;
  o4.y = dy * inv * gg.y + bb.y;
  o4.z = dz * inv * gg.z + bb.z;
  o4.w = dw * inv * gg.w + bb.w;
  *(float4*)&out[(size_t)row * DDIM + t * 4] = o4;
  if (outb) {
    uint2 p; p.x = pack_bf16(o4.x, o4.y); p.y = pack_bf16(o4.z, o4.w);
    *(uint2*)&outb[(size_t)row * DDIM + t * 4] = p;
  }
}

// ---------------------------------------------------------------------------
extern "C" void kernel_launch(void* const* d_in, const int* in_sizes, int n_in,
                              void* d_out, int out_size, void* d_ws, size_t ws_size,
                              hipStream_t stream) {
  const float* x   = (const float*)d_in[0];
  const float* Wq  = (const float*)d_in[1];
  const float* bq  = (const float*)d_in[2];
  const float* Wk  = (const float*)d_in[3];
  const float* bk  = (const float*)d_in[4];
  const float* Wv  = (const float*)d_in[5];
  const float* bv  = (const float*)d_in[6];
  const float* Wo  = (const float*)d_in[7];
  const float* bo  = (const float*)d_in[8];
  const float* g1  = (const float*)d_in[9];
  const float* be1 = (const float*)d_in[10];
  const float* g2  = (const float*)d_in[11];
  const float* be2 = (const float*)d_in[12];
  const float* W1  = (const float*)d_in[13];
  const float* c1  = (const float*)d_in[14];
  const float* W2  = (const float*)d_in[15];
  const float* c2  = (const float*)d_in[16];
  const float* gf  = (const float*)d_in[17];
  const float* bef = (const float*)d_in[18];
  float* out = (float*)d_out;

  // ---- workspace layout ----------------------------------------------------
  const size_t BSD = (size_t)BS * DDIM;                  // 8.39M elements
  float*          h    = (float*)d_ws;                   // BSD fp32
  unsigned short* hb   = (unsigned short*)(h + BSD);     // BSD bf16
  float*          q    = (float*)(hb + BSD);             // BSD fp32
  float*          k    = q + BSD;                        // BSD fp32
  float*          v    = k + BSD;                        // BSD fp32
  unsigned short* attb = (unsigned short*)(v + BSD);     // BSD bf16
  unsigned short* y1b  = (unsigned short*)q;             // BS*F bf16 (reuses q+k)
  float*          kvb  = (float*)(attb + BSD);
  float*          ksb  = kvb + (size_t)BDIM * HDIM * DH * DH;
  unsigned short* wt   = (unsigned short*)(ksb + (size_t)BDIM * HDIM * DH);

  const size_t DD = (size_t)DDIM * DDIM;
  const size_t DF = (size_t)DDIM * FDIM;
  const size_t WL = 4 * DD + 2 * DF;                     // bf16 elems per layer

  const dim3 blk(256);

  // ---- one-shot weight convert + transpose (bf16, [N][K]) ------------------
  for (int l = 0; l < LDIM; ++l) {
    unsigned short* wqt = wt + (size_t)l * WL;
    unsigned short* wkt = wqt + DD;
    unsigned short* wvt = wkt + DD;
    unsigned short* wot = wvt + DD;
    unsigned short* w1t = wot + DD;       // [F][D]
    unsigned short* w2t = w1t + DF;       // [D][F]
    transpose_bf16_kernel<<<dim3(DDIM/32, DDIM/32), blk, 0, stream>>>(Wq + l*DD, wqt, DDIM, DDIM);
    transpose_bf16_kernel<<<dim3(DDIM/32, DDIM/32), blk, 0, stream>>>(Wk + l*DD, wkt, DDIM, DDIM);
    transpose_bf16_kernel<<<dim3(DDIM/32, DDIM/32), blk, 0, stream>>>(Wv + l*DD, wvt, DDIM, DDIM);
    transpose_bf16_kernel<<<dim3(DDIM/32, DDIM/32), blk, 0, stream>>>(Wo + l*DD, wot, DDIM, DDIM);
    transpose_bf16_kernel<<<dim3(FDIM/32, DDIM/32), blk, 0, stream>>>(W1 + l*DF, w1t, DDIM, FDIM);
    transpose_bf16_kernel<<<dim3(DDIM/32, FDIM/32), blk, 0, stream>>>(W2 + l*DF, w2t, FDIM, DDIM);
  }

  dualcast_kernel<<<dim3(BSD / 1024), blk, 0, stream>>>(x, h, hb);

  const dim3 gD(DDIM / 128, BS / 128);     // N=1024 GEMMs
  const dim3 gF(FDIM / 128, BS / 128);     // N=4096 GEMM

  for (int l = 0; l < LDIM; ++l) {
    unsigned short* wqt = wt + (size_t)l * WL;
    unsigned short* wkt = wqt + DD;
    unsigned short* wvt = wkt + DD;
    unsigned short* wot = wvt + DD;
    unsigned short* w1t = wot + DD;
    unsigned short* w2t = w1t + DF;

    // Q,K with phi epilogue; V plain (fp32 outputs for the VALU attention)
    gemm_bf16_wmma<1,0,0><<<gD, blk, 0, stream>>>(q, hb, wqt, bq + l*DDIM, nullptr, BS, DDIM, DDIM);
    gemm_bf16_wmma<1,0,0><<<gD, blk, 0, stream>>>(k, hb, wkt, bk + l*DDIM, nullptr, BS, DDIM, DDIM);
    gemm_bf16_wmma<0,0,0><<<gD, blk, 0, stream>>>(v, hb, wvt, bv + l*DDIM, nullptr, BS, DDIM, DDIM);

    // linear attention state + apply (att written as bf16)
    kv_ksum_kernel<<<dim3(BDIM * HDIM), blk, 0, stream>>>(k, v, kvb, ksb);
    attn_apply_kernel<<<dim3(BDIM * HDIM * (SDIM / 4)), blk, 0, stream>>>(q, kvb, ksb, attb);

    // h = h + att @ Wo + bo ; LN1 in place (regenerates hb)
    gemm_bf16_wmma<0,1,0><<<gD, blk, 0, stream>>>(h, attb, wot, bo + l*DDIM, h, BS, DDIM, DDIM);
    ln_kernel<<<dim3(BS), blk, 0, stream>>>(h, h, g1 + l*DDIM, be1 + l*DDIM, hb);

    // FFN: y1b = bf16(gelu(h@W1 + c1)) ; h = LN2(h + y1b@W2 + c2)
    gemm_bf16_wmma<2,0,1><<<gF, blk, 0, stream>>>(y1b, hb, w1t, c1 + l*FDIM, nullptr, BS, FDIM, DDIM);
    gemm_bf16_wmma<0,1,0><<<gD, blk, 0, stream>>>(h, y1b, w2t, c2 + l*DDIM, h, BS, DDIM, FDIM);
    ln_kernel<<<dim3(BS), blk, 0, stream>>>(h, h, g2 + l*DDIM, be2 + l*DDIM, hb);
  }

  ln_kernel<<<dim3(BS), blk, 0, stream>>>(out, h, gf, bef, nullptr);
}
